// MoE_KAN_MLP_67748814127516
// MI455X (gfx1250) — compile-verified, matrix operand order checked by reading, not run on hardware
//
#include <hip/hip_runtime.h>

typedef __bf16 bf16_t;
typedef __attribute__((ext_vector_type(16))) __bf16 v16bf;
typedef __attribute__((ext_vector_type(8)))  float  v8f;
typedef __attribute__((ext_vector_type(4)))  int    int4v;

static constexpr int T_TOK = 2048;
static constexpr int H_DIM = 1024;
static constexpr int F_DIM = 4096;
static constexpr int E_EXP = 8;
static constexpr int NGRID = 8;
static constexpr int NM    = E_EXP / 2;   // experts per family

#if defined(__HIP_DEVICE_COMPILE__) && __has_builtin(__builtin_amdgcn_global_load_async_to_lds_b128)
#define USE_ASYNC_COPY 1
#else
#define USE_ASYNC_COPY 0
#endif

#if USE_ASYNC_COPY
__device__ __forceinline__ void async_copy_b128(const bf16_t* g, const bf16_t* s) {
  // global (AS1) -> LDS (AS3), 16 bytes, tracked by ASYNCcnt
  __builtin_amdgcn_global_load_async_to_lds_b128(
      (__attribute__((address_space(1))) int4v*)(uintptr_t)g,
      (__attribute__((address_space(3))) int4v*)(uint32_t)(uintptr_t)s,
      0, 0);
}
__device__ __forceinline__ void wait_async0() {
#if __has_builtin(__builtin_amdgcn_s_wait_asynccnt)
  __builtin_amdgcn_s_wait_asynccnt(0);
#else
  asm volatile("s_wait_asynccnt 0x0" ::: "memory");
#endif
}
#endif

// ---------------------------------------------------------------------------
// Router: logits = hs @ gate_w^T ; softmax fp32 ; top-2 ; renormalize -> w[T,E]
// ---------------------------------------------------------------------------
__global__ void router_kernel(const float* __restrict__ hs,
                              const float* __restrict__ gw,
                              float* __restrict__ wout) {
  int t = blockIdx.x * blockDim.x + threadIdx.x;
  if (t >= T_TOK) return;
  float lg[E_EXP];
#pragma unroll
  for (int e = 0; e < E_EXP; ++e) lg[e] = 0.f;
  const float* x = hs + (size_t)t * H_DIM;
  for (int k = 0; k < H_DIM; ++k) {
    float xv = x[k];
#pragma unroll
    for (int e = 0; e < E_EXP; ++e) lg[e] += xv * gw[e * H_DIM + k];
  }
  float mx = lg[0];
#pragma unroll
  for (int e = 1; e < E_EXP; ++e) mx = fmaxf(mx, lg[e]);
  float p[E_EXP]; float sum = 0.f;
#pragma unroll
  for (int e = 0; e < E_EXP; ++e) { p[e] = expf(lg[e] - mx); sum += p[e]; }
#pragma unroll
  for (int e = 0; e < E_EXP; ++e) p[e] /= sum;
  int i0 = 0;
#pragma unroll
  for (int e = 1; e < E_EXP; ++e) if (p[e] > p[i0]) i0 = e;
  int i1 = (i0 == 0) ? 1 : 0;
#pragma unroll
  for (int e = 0; e < E_EXP; ++e) if (e != i0 && p[e] > p[i1]) i1 = e;
  float s = p[i0] + p[i1];
#pragma unroll
  for (int e = 0; e < E_EXP; ++e)
    wout[t * E_EXP + e] = (e == i0) ? p[i0] / s : ((e == i1) ? p[i1] / s : 0.f);
}

// ---------------------------------------------------------------------------
// fp32 -> bf16 bulk convert (8 elems / thread, 16B packed stores)
// ---------------------------------------------------------------------------
__global__ void cvt_f32_bf16(const float* __restrict__ in, bf16_t* __restrict__ out, long n) {
  long i = ((long)blockIdx.x * blockDim.x + threadIdx.x) * 8;
  if (i >= n) return;
  float4 a = *(const float4*)(in + i);
  float4 b = *(const float4*)(in + i + 4);
  union { bf16_t h[8]; uint4 u; } r;
  r.h[0] = (bf16_t)a.x; r.h[1] = (bf16_t)a.y; r.h[2] = (bf16_t)a.z; r.h[3] = (bf16_t)a.w;
  r.h[4] = (bf16_t)b.x; r.h[5] = (bf16_t)b.y; r.h[6] = (bf16_t)b.z; r.h[7] = (bf16_t)b.w;
  *(uint4*)(out + i) = r.u;
}

// ---------------------------------------------------------------------------
// RSWAF basis: x -> 8 bf16 basis values (b = 1 - tanh((x-g)*0.5)^2)
// ---------------------------------------------------------------------------
__device__ __forceinline__ void rswaf8(float x, bf16_t* h) {
#pragma unroll
  for (int g = 0; g < NGRID; ++g) {
    float gv = -1.2f + 0.2f * (float)g;
    float t  = tanhf((x - gv) * 0.5f);
    h[g] = (bf16_t)(1.f - t * t);
  }
}

__global__ void rswaf_kernel(const float* __restrict__ in, bf16_t* __restrict__ out, long n) {
  long i = (long)blockIdx.x * blockDim.x + threadIdx.x;
  if (i >= n) return;
  union { bf16_t h[8]; uint4 u; } r;
  rswaf8(in[i], r.h);
  *(uint4*)(out + i * NGRID) = r.u;
}

// ---------------------------------------------------------------------------
// Tiled bf16 WMMA GEMM: C[M,N] = A[M,K] * B[N,K]^T (both row-major, ld = K)
// 128x128 block tile, BK=64, 8 waves (4x2), wave tile 32x64.
// Double-buffered LDS; async global->LDS copies when available.
// Epilogues: 0 = +bias,GELU->bf16 ; 1 = rswaf x8 ->bf16 ; 2 = out += w[m]*v
// ---------------------------------------------------------------------------
static constexpr int BM = 128, BN = 128, BK = 64;
static constexpr int LDS_S = 72;   // 144B row stride = 36 banks -> conflict-free b128 reads

template <int MODE>
__launch_bounds__(256)
__global__ void gemm_wmma_bf16(const bf16_t* __restrict__ A,
                               const bf16_t* __restrict__ B,
                               int N, int K,
                               void* __restrict__ outp, int ldo,
                               const float* __restrict__ bias,
                               const float* __restrict__ wv, int wcol) {
  __shared__ bf16_t As[2][BM * LDS_S];
  __shared__ bf16_t Bs[2][BN * LDS_S];

  const int tid  = threadIdx.x;
  const int lane = tid & 31;
  const int wid  = tid >> 5;
  const int wm   = wid & 3;        // wave row (4)
  const int wn   = wid >> 2;       // wave col (2)
  const int half = lane >> 4;      // K-half select per ISA A/B fragment layout
  const int lr   = lane & 15;      // row (A) / col (B) within fragment

  const int m0 = blockIdx.y * BM;
  const int n0 = blockIdx.x * BN;

  // cooperative tile copy: 2 threads per row, each 32 elems (4 x b128)
  const int grow = tid >> 1;
  const int gcol = (tid & 1) * 32;

  const bf16_t* AgBase = A + (size_t)(m0 + grow) * K + gcol;
  const bf16_t* BgBase = B + (size_t)(n0 + grow) * K + gcol;

  auto issue_tile = [&](int kb, int buf) {
    const bf16_t* ga = AgBase + kb;
    const bf16_t* gb = BgBase + kb;
    bf16_t* sa = &As[buf][grow * LDS_S + gcol];
    bf16_t* sb = &Bs[buf][grow * LDS_S + gcol];
#if USE_ASYNC_COPY
#pragma unroll
    for (int c = 0; c < 4; ++c) {
      async_copy_b128(ga + c * 8, sa + c * 8);
      async_copy_b128(gb + c * 8, sb + c * 8);
    }
#else
    uint4 ra[4], rb[4];
#pragma unroll
    for (int c = 0; c < 4; ++c) ra[c] = *(const uint4*)(ga + c * 8);
#pragma unroll
    for (int c = 0; c < 4; ++c) rb[c] = *(const uint4*)(gb + c * 8);
#pragma unroll
    for (int c = 0; c < 4; ++c) {
      *(uint4*)(sa + c * 8) = ra[c];
      *(uint4*)(sb + c * 8) = rb[c];
    }
#endif
  };

  v8f zero = {};
  v8f acc[2][4];
#pragma unroll
  for (int i = 0; i < 2; ++i)
#pragma unroll
    for (int j = 0; j < 4; ++j) acc[i][j] = zero;

  const int nk = K / BK;
  issue_tile(0, 0);

  for (int kt = 0; kt < nk; ++kt) {
#if USE_ASYNC_COPY
    wait_async0();          // this wave's copies for tile kt have landed
#endif
    __syncthreads();        // everyone's tile kt in LDS; prev compute done
    if (kt + 1 < nk) issue_tile((kt + 1) * BK, (kt + 1) & 1);  // fill other buffer
    const int buf = kt & 1;

#pragma unroll
    for (int sub = 0; sub < 2; ++sub) {
      const int kof = sub * 32;
      v16bf af[2];
      v16bf bfr[4];
#pragma unroll
      for (int i = 0; i < 2; ++i) {
        int r = wm * 32 + i * 16 + lr;
        uint4* d = (uint4*)&af[i];
        d[0] = *(const uint4*)&As[buf][r * LDS_S + kof + half * 8];
        d[1] = *(const uint4*)&As[buf][r * LDS_S + kof + 16 + half * 8];
      }
#pragma unroll
      for (int j = 0; j < 4; ++j) {
        int r = wn * 64 + j * 16 + lr;
        uint4* d = (uint4*)&bfr[j];
        d[0] = *(const uint4*)&Bs[buf][r * LDS_S + kof + half * 8];
        d[1] = *(const uint4*)&Bs[buf][r * LDS_S + kof + 16 + half * 8];
      }
#pragma unroll
      for (int j = 0; j < 4; ++j)
#pragma unroll
        for (int i = 0; i < 2; ++i)
          acc[i][j] = __builtin_amdgcn_wmma_f32_16x16x32_bf16(
              false, af[i], false, bfr[j], (short)0, acc[i][j], false, false);
    }
  }

  // epilogue: acc[i][j][r] <-> (m = wm*32+i*16+half*8+r, n = wn*64+j*16+lr)
#pragma unroll
  for (int i = 0; i < 2; ++i) {
#pragma unroll
    for (int j = 0; j < 4; ++j) {
#pragma unroll
      for (int r = 0; r < 8; ++r) {
        int m = m0 + wm * 32 + i * 16 + half * 8 + r;
        int n = n0 + wn * 64 + j * 16 + lr;
        float v = acc[i][j][r];
        if (MODE == 0) {
          v += bias ? bias[n] : 0.f;
          float gl = 0.5f * v * (1.f + erff(v * 0.70710678118654752f));
          ((bf16_t*)outp)[(size_t)m * ldo + n] = (bf16_t)gl;
        } else if (MODE == 1) {
          union { bf16_t h[8]; uint4 u; } rr;
          rswaf8(v, rr.h);
          *(uint4*)&((bf16_t*)outp)[(size_t)m * ldo + (size_t)n * NGRID] = rr.u;
        } else {
          v += bias ? bias[n] : 0.f;
          float s = wv[m * E_EXP + wcol];
          float* o = (float*)outp + (size_t)m * ldo + n;
          *o += s * v;   // unique owner per element; kernels serialized -> deterministic
        }
      }
    }
  }
}

// ---------------------------------------------------------------------------
// Launch
// inputs: hidden_states, gate_w, fc1_w, fc1_b, fc2_w, fc2_b, sp1_w, sp2_w (fp32)
// ---------------------------------------------------------------------------
extern "C" void kernel_launch(void* const* d_in, const int* in_sizes, int n_in,
                              void* d_out, int out_size, void* d_ws, size_t ws_size,
                              hipStream_t stream) {
  const float* hs   = (const float*)d_in[0];
  const float* gate = (const float*)d_in[1];
  const float* fc1w = (const float*)d_in[2];
  const float* fc1b = (const float*)d_in[3];
  const float* fc2w = (const float*)d_in[4];
  const float* fc2b = (const float*)d_in[5];
  const float* sp1w = (const float*)d_in[6];
  const float* sp2w = (const float*)d_in[7];
  float* out = (float*)d_out;

  char* ws = (char*)d_ws;
  auto alloc = [&](size_t bytes) {
    char* p = ws;
    ws += (bytes + 255) & ~size_t(255);
    return (void*)p;
  };
  float*  wv   = (float*) alloc((size_t)T_TOK * E_EXP * 4);                // router weights
  bf16_t* hsb  = (bf16_t*)alloc((size_t)T_TOK * H_DIM * 2);                // hs bf16
  bf16_t* b1   = (bf16_t*)alloc((size_t)T_TOK * H_DIM * NGRID * 2);       // rswaf(hs)
  bf16_t* hbuf = (bf16_t*)alloc((size_t)T_TOK * F_DIM * 2);                // gelu(fc1) per expert
  bf16_t* b2   = (bf16_t*)alloc((size_t)T_TOK * (F_DIM / 2) * NGRID * 2); // rswaf(k1) per expert
  bf16_t* wA   = (bf16_t*)alloc((size_t)(F_DIM / 2) * H_DIM * NGRID * 2); // max fc1/sp1 per-expert
  bf16_t* wB   = (bf16_t*)alloc((size_t)H_DIM * (F_DIM / 2) * NGRID * 2); // max fc2/sp2 per-expert

  (void)hipMemsetAsync(d_out, 0, (size_t)T_TOK * H_DIM * sizeof(float), stream);

  router_kernel<<<T_TOK / 256, 256, 0, stream>>>(hs, gate, wv);
  {
    long n = (long)T_TOK * H_DIM;
    cvt_f32_bf16<<<(unsigned)((n / 8 + 255) / 256), 256, 0, stream>>>(hs, hsb, n);
    rswaf_kernel<<<(unsigned)((n + 255) / 256), 256, 0, stream>>>(hs, b1, n);
  }

  // ---- MLP experts ----
  for (int e = 0; e < NM; ++e) {
    long n1 = (long)F_DIM * H_DIM;
    cvt_f32_bf16<<<(unsigned)((n1 / 8 + 255) / 256), 256, 0, stream>>>(fc1w + (size_t)e * n1, wA, n1);
    gemm_wmma_bf16<0><<<dim3(F_DIM / BN, T_TOK / BM), 256, 0, stream>>>(
        hsb, wA, F_DIM, H_DIM, hbuf, F_DIM, fc1b + (size_t)e * F_DIM, nullptr, 0);

    long n2 = (long)H_DIM * F_DIM;
    cvt_f32_bf16<<<(unsigned)((n2 / 8 + 255) / 256), 256, 0, stream>>>(fc2w + (size_t)e * n2, wB, n2);
    gemm_wmma_bf16<2><<<dim3(H_DIM / BN, T_TOK / BM), 256, 0, stream>>>(
        hbuf, wB, H_DIM, F_DIM, out, H_DIM, fc2b + (size_t)e * H_DIM, wv, e);
  }

  // ---- FasterKAN experts ----
  for (int e = 0; e < NM; ++e) {
    long n1 = (long)(F_DIM / 2) * (H_DIM * NGRID);
    cvt_f32_bf16<<<(unsigned)((n1 / 8 + 255) / 256), 256, 0, stream>>>(sp1w + (size_t)e * n1, wA, n1);
    gemm_wmma_bf16<1><<<dim3((F_DIM / 2) / BN, T_TOK / BM), 256, 0, stream>>>(
        b1, wA, F_DIM / 2, H_DIM * NGRID, b2, (F_DIM / 2) * NGRID, nullptr, nullptr, 0);

    long n2 = (long)H_DIM * ((F_DIM / 2) * NGRID);
    cvt_f32_bf16<<<(unsigned)((n2 / 8 + 255) / 256), 256, 0, stream>>>(sp2w + (size_t)e * n2, wB, n2);
    gemm_wmma_bf16<2><<<dim3(H_DIM / BN, T_TOK / BM), 256, 0, stream>>>(
        b2, wB, H_DIM, (F_DIM / 2) * NGRID, out, H_DIM, nullptr, wv, NM + e);
  }
}
